// DownBlock_17222818857411
// MI455X (gfx1250) — 8-bit matrix operand order repaired and checked by reading, compiled, not run on hardware
//
#include <hip/hip_runtime.h>
#include <cstdint>
#include <cstddef>


typedef __attribute__((ext_vector_type(8))) int v8i;

#define PCH   256
#define HW    3136
#define WID   56
#define NIMG  16
#define EPS   1e-5f

static constexpr size_t OFF_ACT1   = 0;
static constexpr size_t OFF_SIGN2  = 12845056;
static constexpr size_t OFF_RES1   = 38535168;
static constexpr size_t OFF_WQ1    = 89915392;
static constexpr size_t OFF_WQ2    = 90505216;
static constexpr size_t OFF_ALPHA1 = 91684864;
static constexpr size_t OFF_ADD1   = OFF_ALPHA1 + 1024;
static constexpr size_t OFF_ALPHA2 = OFF_ALPHA1 + 2048;
static constexpr size_t OFF_ADD2   = OFF_ALPHA1 + 3072;

union V8U { v8i v; int4 q[2]; };

__global__ void prep_scales(const float* __restrict__ w1, const float* __restrict__ w2,
                            const float* g1, const float* b1, const float* m1, const float* v1,
                            const float* g2, const float* b2, const float* m2, const float* v2,
                            float* alpha1, float* add1, float* alpha2, float* add2) {
    int t = blockIdx.x * blockDim.x + threadIdx.x;
    if (t < 256) {
        int co = t;
        float s = 0.f;
        const float* wp = w1 + (size_t)co * (256 * 9);
        for (int i = 0; i < 256 * 9; ++i) s += fabsf(wp[i]);
        s *= (1.f / (256.f * 9.f));
        float inv = g1[co] * rsqrtf(v1[co] + EPS);
        alpha1[co] = s * inv;
        add1[co]   = b1[co] - m1[co] * inv;
    } else if (t < 512) {
        int co = t - 256;
        float s = 0.f;
        const float* wp = w2 + (size_t)co * (512 * 9);
        for (int i = 0; i < 512 * 9; ++i) s += fabsf(wp[i]);
        s *= (1.f / (512.f * 9.f));
        float inv = g2[co] * rsqrtf(v2[co] + EPS);
        alpha2[co] = s * inv;
        add2[co]   = b2[co] - m2[co] * inv;
    }
}

template <int CIN>
__global__ void pack_weights(const float* __restrict__ w, int8_t* __restrict__ wq) {
    constexpr int KC = CIN / 64;
    int idx    = blockIdx.x * 256 + threadIdx.x;
    int b      = idx & 31;
    int lane   = (idx >> 5) & 31;
    int kc     = (idx >> 10) % KC;
    int kpos   = (idx / (1024 * KC)) % 9;
    int coTile = idx / (1024 * KC * 9);
    int co = coTile * 16 + (lane & 15);
    int hi = lane >> 4;
    int K  = ((b >> 4) << 5) + (hi << 4) + (b & 15);
    int ci = kc * 64 + K;
    int kh = kpos / 3, kw = kpos % 3;
    float wv = w[(((size_t)co * CIN + ci) * 3 + kh) * 3 + kw];
    wq[idx] = (wv >= 0.f) ? (int8_t)1 : (int8_t)-1;
}

__global__ void binarize_act(const float* __restrict__ x, int8_t* __restrict__ act8) {
    int idx = blockIdx.x * 256 + threadIdx.x;
    int c  = idx & 255;
    int p  = (idx >> 8) % HW;
    int nI = idx / (HW * 256);
    float v = x[((size_t)nI * 512 + c) * HW + p];
    act8[idx] = (v >= 0.f) ? (int8_t)1 : (int8_t)-1;
}

__global__ void idle_prep(const float* __restrict__ x, const float* __restrict__ mb,
                          int8_t* __restrict__ sign2, float* __restrict__ res1) {
    int idx = blockIdx.x * 256 + threadIdx.x;
    int c  = idx & 255;
    int p  = (idx >> 8) % HW;
    int nI = idx / (HW * 256);
    float t = x[((size_t)nI * 512 + 256 + c) * HW + p] + mb[c];
    sign2[((size_t)nI * HW + p) * 512 + 2 * c + 1] = (t >= 0.f) ? (int8_t)1 : (int8_t)-1;
    if (c < 128) res1[((size_t)nI * 256 + 2 * c + 1) * HW + p] = t;
}

template <int CIN>
__device__ __forceinline__ void bin_conv_tile4(const int8_t* __restrict__ act,
                                               const int8_t* __restrict__ wq,
                                               int nImg, int p0, int coTile,
                                               int4* ldsB  ,
                                               v8i acc[4]) {
    constexpr int KC  = CIN / 64;
    constexpr int NIT = 9 * KC;
    const int tid  = threadIdx.x;
    const int lane = tid & 31;

    const int col  = tid >> 2;
    const int ck   = tid & 3;
    const int p    = p0 + col;
    const int h    = p / WID;
    const int w    = p % WID;
    const int dstIdx = (col >> 4) * 64 + ((col & 15) + ((ck & 1) << 4)) * 2 + (ck >> 1);
    const int kbyte  = ck * 16;
    const size_t rowBase = (size_t)nImg * HW;

    auto stage = [&](int it, int buf) {
        int kpos = it / KC;
        int kc   = it - kpos * KC;
        int dh   = kpos / 3 - 1;
        int dw   = kpos - (kpos / 3) * 3 - 1;
        int hh = h + dh, ww = w + dw;
        int4 v; v.x = 0; v.y = 0; v.z = 0; v.w = 0;
        if (((unsigned)hh < WID) & ((unsigned)ww < WID)) {
            size_t off = (rowBase + (size_t)(hh * WID + ww)) * CIN + kc * 64 + kbyte;
            v = *(const int4*)(act + off);
        }
        ldsB[buf * 256 + dstIdx] = v;
    };

    v8i z = {};
    acc[0] = z; acc[1] = z; acc[2] = z; acc[3] = z;

    stage(0, 0);
    __syncthreads();
    const int8_t* wbase = wq + (size_t)coTile * 9 * KC * 1024 + lane * 32;
    for (int it = 0; it < NIT; ++it) {
        int cur = it & 1;
        if (it + 1 < NIT) {
            stage(it + 1, cur ^ 1);
            __builtin_prefetch(wbase + (size_t)(it + 1) * 1024, 0, 1);
        }
        V8U a;
        const int4* ap = (const int4*)(wbase + (size_t)it * 1024);
        a.q[0] = ap[0]; a.q[1] = ap[1];
#pragma unroll
        for (int s = 0; s < 4; ++s) {
            V8U b;
            b.q[0] = ldsB[cur * 256 + s * 64 + lane * 2];
            b.q[1] = ldsB[cur * 256 + s * 64 + lane * 2 + 1];
            acc[s] = __builtin_amdgcn_wmma_i32_16x16x64_iu8(true, a.v, true, b.v,
                                                            acc[s], false, false);
        }
        __syncthreads();
    }
}

__global__ void __launch_bounds__(256)
conv1_kernel(const int8_t* __restrict__ act1, const int8_t* __restrict__ wq1,
             const float* __restrict__ x, const float* __restrict__ alpha1,
             const float* __restrict__ add1,
             int8_t* __restrict__ sign2, float* __restrict__ res1) {
    __shared__ __align__(32) int4 ldsB[512];
    int p0     = blockIdx.x * 64;
    int nImg   = blockIdx.y;
    int waveId = threadIdx.x >> 5, lane = threadIdx.x & 31;
    int coTile = blockIdx.z * 8 + waveId;

    v8i acc[4];
    bin_conv_tile4<256>(act1, wq1, nImg, p0, coTile, ldsB, acc);

    int rowh = lane >> 4;
    float al[8], ad[8];
#pragma unroll
    for (int r = 0; r < 8; ++r) {
        int co = coTile * 16 + r + 8 * rowh;
        al[r] = alpha1[co];
        ad[r] = add1[co];
    }
#pragma unroll
    for (int s = 0; s < 4; ++s) {
        int pix = p0 + s * 16 + (lane & 15);
#pragma unroll
        for (int r = 0; r < 8; ++r) {
            int co = coTile * 16 + r + 8 * rowh;
            float val = (float)acc[s][r] * al[r] + ad[r]
                      + x[((size_t)nImg * 512 + co) * HW + pix];
            val = fminf(fmaxf(val, -1.f), 1.f);
            sign2[((size_t)nImg * HW + pix) * 512 + 2 * co] =
                (val >= 0.f) ? (int8_t)1 : (int8_t)-1;
            if (co < 128)
                res1[((size_t)nImg * 256 + 2 * co) * HW + pix] = val;
        }
    }
}

__global__ void __launch_bounds__(256)
conv2_kernel(const int8_t* __restrict__ sign2, const int8_t* __restrict__ wq2,
             const float* __restrict__ res1, const float* __restrict__ alpha2,
             const float* __restrict__ add2, float* __restrict__ out) {
    __shared__ __align__(32) int4 ldsB[512];
    int p0     = blockIdx.x * 64;
    int nImg   = blockIdx.y;
    int waveId = threadIdx.x >> 5, lane = threadIdx.x & 31;
    int coTile = blockIdx.z * 8 + waveId;

    v8i acc[4];
    bin_conv_tile4<512>(sign2, wq2, nImg, p0, coTile, ldsB, acc);

    int rowh = lane >> 4;
    float al[8], ad[8];
#pragma unroll
    for (int r = 0; r < 8; ++r) {
        int co = coTile * 16 + r + 8 * rowh;
        al[r] = alpha2[co];
        ad[r] = add2[co];
    }
#pragma unroll
    for (int s = 0; s < 4; ++s) {
        int pix = p0 + s * 16 + (lane & 15);
#pragma unroll
        for (int r = 0; r < 8; ++r) {
            int co = coTile * 16 + r + 8 * rowh;
            float val = (float)acc[s][r] * al[r] + ad[r]
                      + res1[((size_t)nImg * 256 + co) * HW + pix];
            val = fminf(fmaxf(val, -1.f), 1.f);
            out[((size_t)nImg * 256 + co) * HW + pix] = val;
        }
    }
}

extern "C" void kernel_launch(void* const* d_in, const int* in_sizes, int n_in,
                              void* d_out, int out_size, void* d_ws, size_t ws_size,
                              hipStream_t stream) {
    const float* x      = (const float*)d_in[0];
    const float* w1     = (const float*)d_in[1];
    const float* w2     = (const float*)d_in[2];
    const float* gamma1 = (const float*)d_in[3];
    const float* beta1  = (const float*)d_in[4];
    const float* mean1  = (const float*)d_in[5];
    const float* var1   = (const float*)d_in[6];
    const float* gamma2 = (const float*)d_in[7];
    const float* beta2  = (const float*)d_in[8];
    const float* mean2  = (const float*)d_in[9];
    const float* var2   = (const float*)d_in[10];
    const float* mbias  = (const float*)d_in[11];

    char* ws = (char*)d_ws;
    int8_t* act1   = (int8_t*)(ws + OFF_ACT1);
    int8_t* sign2  = (int8_t*)(ws + OFF_SIGN2);
    float*  res1   = (float*) (ws + OFF_RES1);
    int8_t* wq1    = (int8_t*)(ws + OFF_WQ1);
    int8_t* wq2    = (int8_t*)(ws + OFF_WQ2);
    float*  alpha1 = (float*) (ws + OFF_ALPHA1);
    float*  add1   = (float*) (ws + OFF_ADD1);
    float*  alpha2 = (float*) (ws + OFF_ALPHA2);
    float*  add2   = (float*) (ws + OFF_ADD2);
    float*  out    = (float*)d_out;

    prep_scales<<<2, 256, 0, stream>>>(w1, w2, gamma1, beta1, mean1, var1,
                                       gamma2, beta2, mean2, var2,
                                       alpha1, add1, alpha2, add2);
    pack_weights<256><<<2304, 256, 0, stream>>>(w1, wq1);
    pack_weights<512><<<4608, 256, 0, stream>>>(w2, wq2);
    binarize_act<<<50176, 256, 0, stream>>>(x, act1);
    idle_prep<<<50176, 256, 0, stream>>>(x, mbias, sign2, res1);

    conv1_kernel<<<dim3(49, NIMG, 2), 256, 0, stream>>>(act1, wq1, x, alpha1, add1,
                                                        sign2, res1);
    conv2_kernel<<<dim3(49, NIMG, 2), 256, 0, stream>>>(sign2, wq2, res1, alpha2, add2,
                                                        out);
}
